// SpatialAttention_10033043604186
// MI455X (gfx1250) — compile-verified, matrix operand order checked by reading, not run on hardware
//
#include <hip/hip_runtime.h>

typedef __attribute__((ext_vector_type(16))) _Float16 v16h;
typedef __attribute__((ext_vector_type(8)))  _Float16 v8h;
typedef __attribute__((ext_vector_type(8)))  float    v8f;
typedef __attribute__((ext_vector_type(4)))  float    v4f;

#define D_MODEL   768
#define NUM_HEADS 12
#define HEAD_DIM  64
#define SEQ       1024
#define BATCH     4
#define M_ROWS    (BATCH * SEQ)      // 4096

// Build a 16x32 / 32x16 f16 WMMA fragment from two 8-half chunks.
static __device__ __forceinline__ v16h load_h16(const _Float16* p0, const _Float16* p1) {
  v8h a0 = *(const v8h*)p0;
  v8h a1 = *(const v8h*)p1;
  v16h r;
#pragma unroll
  for (int i = 0; i < 8; ++i) { r[i] = a0[i]; r[8 + i] = a1[i]; }
  return r;
}

static __device__ __forceinline__ v8f wmma_f16(v16h a, v16h b, v8f c) {
  return __builtin_amdgcn_wmma_f32_16x16x32_f16(false, a, false, b, (short)0, c, false, false);
}

// -------- One-shot f32 -> f16 conversion (memory-bound, runs once) --------
__global__ __launch_bounds__(256)
void cvt_f16_kernel(const float* __restrict__ in, _Float16* __restrict__ out, int n8) {
  const int i = blockIdx.x * blockDim.x + threadIdx.x;
  if (i >= n8) return;
  const v4f x0 = *(const v4f*)(in + i * 8);
  const v4f x1 = *(const v4f*)(in + i * 8 + 4);
  v8h r;
#pragma unroll
  for (int j = 0; j < 4; ++j) { r[j] = (_Float16)x0[j]; r[4 + j] = (_Float16)x1[j]; }
  *(v8h*)(out + i * 8) = r;
}

// -------- QKV projection: out = (X @ W^T + bias) * scale (all-f16 GEMM) ----
// Each wave computes a 16x64 strip: 4 accumulators share one A-fragment.
// transposed==0 : out[b,h,s,d]   (Q, K)
// transposed==1 : out[b,h,d,s]   (V transposed for contiguous PV fragments)
__global__ __launch_bounds__(128)
void proj_qkv_kernel(const _Float16* __restrict__ X, const _Float16* __restrict__ W,
                     const float* __restrict__ bias, _Float16* __restrict__ out,
                     float scale, int transposed) {
  const int wave = blockIdx.x * (blockDim.x >> 5) + (threadIdx.x >> 5);
  const int lane = threadIdx.x & 31;
  const int lr = lane & 15, hi = lane >> 4;
  const int NS = D_MODEL / 64;                 // 12 column strips
  const int mt = wave / NS, ns = wave % NS;

  const _Float16* pa  = X + (mt * 16 + lr) * D_MODEL + hi * 8;   // A: lane = row
  const _Float16* pb0 = W + (ns * 64 + lr) * D_MODEL + hi * 8;   // B: lane = col

  v8f acc[4] = {};
#pragma unroll 2
  for (int kk = 0; kk < D_MODEL; kk += 32) {
    const v16h a = load_h16(pa + kk, pa + kk + 16);
#pragma unroll
    for (int j = 0; j < 4; ++j) {
      const _Float16* pb = pb0 + j * 16 * D_MODEL;
      const v16h b = load_h16(pb + kk, pb + kk + 16);
      acc[j] = wmma_f16(a, b, acc[j]);
    }
  }

#pragma unroll
  for (int j = 0; j < 4; ++j) {
    const int ncol = ns * 64 + j * 16 + lr;
    const int h = ncol >> 6, d = ncol & 63;
    const float bv = bias[ncol];
#pragma unroll
    for (int r = 0; r < 8; ++r) {
      const int m  = mt * 16 + r + 8 * hi;     // C/D layout: half-wave row split
      const int bi = m >> 10, s = m & (SEQ - 1);
      const float val = (acc[j][r] + bv) * scale;
      int idx;
      if (!transposed) idx = ((bi * NUM_HEADS + h) * SEQ + s) * HEAD_DIM + d;
      else             idx = ((bi * NUM_HEADS + h) * HEAD_DIM + d) * SEQ + s;
      out[idx] = (_Float16)val;
    }
  }
}

// -------- Flash attention with analytic spatial bias --------
__global__ __launch_bounds__(256)
void attn_kernel(const _Float16* __restrict__ Q, const _Float16* __restrict__ Kh,
                 const _Float16* __restrict__ Vt, const float* __restrict__ boxes,
                 const float* __restrict__ sbias, _Float16* __restrict__ attended) {
  __shared__ _Float16 ldsP[8 * 16 * 32];       // 1 KB per wave, 8 waves/block
  const int wid  = threadIdx.x >> 5;
  const int lane = threadIdx.x & 31;
  const int lr = lane & 15, hi = lane >> 4;
  const int gw = blockIdx.x * 8 + wid;
  const int qt = gw & 63;
  const int h  = (gw >> 6) % NUM_HEADS;
  const int b  = gw / (64 * NUM_HEADS);
  const int qbase = qt * 16;
  _Float16* lds = ldsP + wid * 512;

  const _Float16* qhead = Q  + (b * NUM_HEADS + h) * SEQ * HEAD_DIM;
  const _Float16* khead = Kh + (b * NUM_HEADS + h) * SEQ * HEAD_DIM;
  const _Float16* vhead = Vt + (b * NUM_HEADS + h) * HEAD_DIM * SEQ;
  const float* boxb = boxes + b * SEQ * 4;
  const float sb = sbias[h];

  // Loop-invariant Q A-fragments (Q already scaled by 1/sqrt(Dh)).
  const _Float16* qrow = qhead + (qbase + lr) * HEAD_DIM + hi * 8;
  const v16h aq0 = load_h16(qrow,      qrow + 16);   // d = 0..31
  const v16h aq1 = load_h16(qrow + 32, qrow + 48);   // d = 32..63

  // Query box centers for this lane's 8 rows.
  float qcx[8], qcy[8];
#pragma unroll
  for (int r = 0; r < 8; ++r) {
    const v4f bx = *(const v4f*)(boxb + (qbase + r + 8 * hi) * 4);
    qcx[r] = (bx[0] + bx[2]) * 0.5f;
    qcy[r] = (bx[1] + bx[3]) * 0.5f;
  }

  v8f o0 = {}, o1 = {}, o2 = {}, o3 = {};
  float mrun[8], lrun[8];
#pragma unroll
  for (int r = 0; r < 8; ++r) { mrun[r] = -3.0e38f; lrun[r] = 0.0f; }

  for (int kk = 0; kk < SEQ; kk += 32) {
    // ---- scores: Q (16xDh) x K^T (Dhx32) -> two 16x16 f32 tiles ----
    const _Float16* krow0 = khead + (kk + lr) * HEAD_DIM + hi * 8;
    const _Float16* krow1 = krow0 + 16 * HEAD_DIM;
    v16h b00 = load_h16(krow0,      krow0 + 16);
    v16h b01 = load_h16(krow0 + 32, krow0 + 48);
    v16h b10 = load_h16(krow1,      krow1 + 16);
    v16h b11 = load_h16(krow1 + 32, krow1 + 48);
    v8f s0 = {}, s1 = {};
    s0 = wmma_f16(aq0, b00, s0); s0 = wmma_f16(aq1, b01, s0);
    s1 = wmma_f16(aq0, b10, s1); s1 = wmma_f16(aq1, b11, s1);

    // key centers for this lane's two columns
    const v4f bk0 = *(const v4f*)(boxb + (kk + lr) * 4);
    const v4f bk1 = *(const v4f*)(boxb + (kk + 16 + lr) * 4);
    const float kc0x = (bk0[0] + bk0[2]) * 0.5f, kc0y = (bk0[1] + bk0[3]) * 0.5f;
    const float kc1x = (bk1[0] + bk1[2]) * 0.5f, kc1y = (bk1[1] + bk1[3]) * 0.5f;

    // ---- bias + online softmax (rows live in 16-lane half-groups) ----
#pragma unroll
    for (int r = 0; r < 8; ++r) {
      float dx0 = qcx[r] - kc0x, dy0 = qcy[r] - kc0y;
      float dx1 = qcx[r] - kc1x, dy1 = qcy[r] - kc1y;
      s0[r] -= sqrtf(dx0 * dx0 + dy0 * dy0) * sb;
      s1[r] -= sqrtf(dx1 * dx1 + dy1 * dy1) * sb;

      float t = fmaxf(s0[r], s1[r]);
#pragma unroll
      for (int off = 1; off < 16; off <<= 1) t = fmaxf(t, __shfl_xor(t, off, 16));
      const float mnew  = fmaxf(mrun[r], t);
      const float alpha = __expf(mrun[r] - mnew);
      const float p0 = __expf(s0[r] - mnew);
      const float p1 = __expf(s1[r] - mnew);
      float rs = p0 + p1;
#pragma unroll
      for (int off = 1; off < 16; off <<= 1) rs += __shfl_xor(rs, off, 16);
      lrun[r] = lrun[r] * alpha + rs;
      mrun[r] = mnew;
      o0[r] *= alpha; o1[r] *= alpha; o2[r] *= alpha; o3[r] *= alpha;

      // store probabilities (C-layout) to LDS as a plain 16x32 tile
      lds[(r + 8 * hi) * 32 + lr]      = (_Float16)p0;
      lds[(r + 8 * hi) * 32 + 16 + lr] = (_Float16)p1;
    }

    asm volatile("s_wait_dscnt 0" ::: "memory");   // wave-local LDS RAW

    // re-read P in A-fragment layout (16x32, K = 32 keys)
    const _Float16* prow = lds + lr * 32 + hi * 8;
    v16h pa = load_h16(prow, prow + 16);

    // ---- P (16x32) x V (32x64), V transposed so key dim is contiguous ----
#pragma unroll
    for (int j = 0; j < 4; ++j) {
      const _Float16* vp = vhead + (j * 16 + lr) * SEQ + kk + hi * 8;
      v16h vb = load_h16(vp, vp + 16);
      v8f* oj = (j == 0) ? &o0 : (j == 1) ? &o1 : (j == 2) ? &o2 : &o3;
      *oj = wmma_f16(pa, vb, *oj);
    }
  }

  // normalize and store attended as f16 [B, S, H*Dh] row-major
#pragma unroll
  for (int r = 0; r < 8; ++r) {
    const float inv = 1.0f / lrun[r];
    const int srow = qbase + r + 8 * hi;
    _Float16* op = attended + (b * SEQ + srow) * D_MODEL + h * HEAD_DIM;
    op[0 * 16 + lr] = (_Float16)(o0[r] * inv);
    op[1 * 16 + lr] = (_Float16)(o1[r] * inv);
    op[2 * 16 + lr] = (_Float16)(o2[r] * inv);
    op[3 * 16 + lr] = (_Float16)(o3[r] * inv);
  }
}

// -------- Output projection: f32 out = attended(f16) @ Wo^T(f16) + bo ------
__global__ __launch_bounds__(128)
void out_proj_kernel(const _Float16* __restrict__ A, const _Float16* __restrict__ W,
                     const float* __restrict__ bias, float* __restrict__ out) {
  const int wave = blockIdx.x * (blockDim.x >> 5) + (threadIdx.x >> 5);
  const int lane = threadIdx.x & 31;
  const int lr = lane & 15, hi = lane >> 4;
  const int NS = D_MODEL / 64;
  const int mt = wave / NS, ns = wave % NS;

  const _Float16* pa  = A + (mt * 16 + lr) * D_MODEL + hi * 8;
  const _Float16* pb0 = W + (ns * 64 + lr) * D_MODEL + hi * 8;

  v8f acc[4] = {};
#pragma unroll 2
  for (int kk = 0; kk < D_MODEL; kk += 32) {
    const v16h a = load_h16(pa + kk, pa + kk + 16);
#pragma unroll
    for (int j = 0; j < 4; ++j) {
      const _Float16* pb = pb0 + j * 16 * D_MODEL;
      const v16h b = load_h16(pb + kk, pb + kk + 16);
      acc[j] = wmma_f16(a, b, acc[j]);
    }
  }

#pragma unroll
  for (int j = 0; j < 4; ++j) {
    const int ncol = ns * 64 + j * 16 + lr;
    const float bv = bias[ncol];
#pragma unroll
    for (int r = 0; r < 8; ++r) {
      const int m = mt * 16 + r + 8 * hi;
      out[m * D_MODEL + ncol] = acc[j][r] + bv;
    }
  }
}

extern "C" void kernel_launch(void* const* d_in, const int* in_sizes, int n_in,
                              void* d_out, int out_size, void* d_ws, size_t ws_size,
                              hipStream_t stream) {
  (void)in_sizes; (void)n_in; (void)out_size; (void)ws_size;
  const float* x     = (const float*)d_in[0];
  const float* boxes = (const float*)d_in[1];
  const float* Wq = (const float*)d_in[2];  const float* bq = (const float*)d_in[3];
  const float* Wk = (const float*)d_in[4];  const float* bk = (const float*)d_in[5];
  const float* Wv = (const float*)d_in[6];  const float* bv = (const float*)d_in[7];
  const float* Wo = (const float*)d_in[8];  const float* bo = (const float*)d_in[9];
  const float* sb = (const float*)d_in[10];
  float* out = (float*)d_out;

  const size_t xe = (size_t)M_ROWS * D_MODEL;   // 3,145,728 halves
  const size_t we = (size_t)D_MODEL * D_MODEL;  //   589,824 halves
  _Float16* Qh  = (_Float16*)d_ws;
  _Float16* Khb = Qh  + xe;
  _Float16* Vth = Khb + xe;
  _Float16* Ah  = Vth + xe;
  _Float16* Xh  = Ah  + xe;
  _Float16* Wqh = Xh  + xe;
  _Float16* Wkh = Wqh + we;
  _Float16* Wvh = Wkh + we;
  _Float16* Woh = Wvh + we;                     // ~34.5 MB total workspace

  // One-shot f32 -> f16 conversion of activations and weights.
  cvt_f16_kernel<<<dim3((int)(xe / 8 / 256)), dim3(256), 0, stream>>>(x,  Xh,  (int)(xe / 8));
  cvt_f16_kernel<<<dim3((int)(we / 8 / 256)), dim3(256), 0, stream>>>(Wq, Wqh, (int)(we / 8));
  cvt_f16_kernel<<<dim3((int)(we / 8 / 256)), dim3(256), 0, stream>>>(Wk, Wkh, (int)(we / 8));
  cvt_f16_kernel<<<dim3((int)(we / 8 / 256)), dim3(256), 0, stream>>>(Wv, Wvh, (int)(we / 8));
  cvt_f16_kernel<<<dim3((int)(we / 8 / 256)), dim3(256), 0, stream>>>(Wo, Woh, (int)(we / 8));

  // 256 M-tiles x 12 N-strips = 3072 waves, 4 waves/block.
  dim3 blk(128), grd((M_ROWS / 16) * (D_MODEL / 64) / 4);
  proj_qkv_kernel<<<grd, blk, 0, stream>>>(Xh, Wqh, bq, Qh,  0.125f, 0); // 1/sqrt(64)
  proj_qkv_kernel<<<grd, blk, 0, stream>>>(Xh, Wkh, bk, Khb, 1.0f,   0);
  proj_qkv_kernel<<<grd, blk, 0, stream>>>(Xh, Wvh, bv, Vth, 1.0f,   1);

  attn_kernel<<<dim3(BATCH * NUM_HEADS * (SEQ / 16) / 8), dim3(256), 0, stream>>>(
      Qh, Khb, Vth, boxes, sb, Ah);

  out_proj_kernel<<<grd, blk, 0, stream>>>(Ah, Woh, bo, out);
}
